// SlidingWindowAttention_73589969650216
// MI455X (gfx1250) — compile-verified, matrix operand order checked by reading, not run on hardware
//
#include <hip/hip_runtime.h>

typedef __attribute__((ext_vector_type(16))) _Float16 v16h;
typedef __attribute__((ext_vector_type(8)))  _Float16 v8h;
typedef __attribute__((ext_vector_type(8)))  float    v8f;
typedef __attribute__((ext_vector_type(4)))  float    v4f;

#define DMODEL 256
#define TSEQ   2048
#define BATCH  2
#define WIN    128
#define HALF_W 64
#define SPAN   160              // keys covering 16 queries: [t0-64, t0+15+63] = 143 -> pad 160
#define NROWS  (BATCH * TSEQ)   // 4096

__device__ __forceinline__ int clampi(int v, int lo, int hi) {
  return v < lo ? lo : (v > hi ? hi : v);
}

// Load one 16x32 f16 WMMA fragment (A or B) from a row-major source where the
// contraction dim K is contiguous. Caller passes this lane's row pointer
// (A: row m = lane&15, B: row n = lane&15). Per ISA layout, lanes 0-15 hold
// K = k0+{0..7, 16..23}, lanes 16-31 hold K = k0+{8..15, 24..31}.
__device__ __forceinline__ v16h load_frag_row(const _Float16* rowPtr, int k0) {
  int kh = ((threadIdx.x & 31) >> 4) * 8;
  const _Float16* p = rowPtr + k0 + kh;
  v8h lo = *(const v8h*)p;
  v8h hi = *(const v8h*)(p + 16);
  v16h a;
#pragma unroll
  for (int i = 0; i < 8; ++i) { a[i] = lo[i]; a[i + 8] = hi[i]; }
  return a;
}

__device__ __forceinline__ v8f wmma16x16x32(v16h a, v16h b, v8f c) {
  return __builtin_amdgcn_wmma_f32_16x16x32_f16(false, a, false, b, (short)0, c,
                                                false, false);
}

// ---------------------------------------------------------------------------
// K0: cast + transpose all four weight matrices to f16, N-major (K contiguous)
// ---------------------------------------------------------------------------
__global__ void prep_weights_kernel(const float* __restrict__ Wq,
                                    const float* __restrict__ Wk,
                                    const float* __restrict__ Wv,
                                    const float* __restrict__ Wo,
                                    _Float16* WqT, _Float16* WkT,
                                    _Float16* WvT, _Float16* WoT) {
  int tid = blockIdx.x * blockDim.x + threadIdx.x;   // 4 * 65536 threads
  int m = tid >> 16;
  int e = tid & 0xFFFF;
  int n = e >> 8;
  int k = e & 255;
  const float* src = (m == 0) ? Wq : (m == 1) ? Wk : (m == 2) ? Wv : Wo;
  _Float16* dst = (m == 0) ? WqT : (m == 1) ? WkT : (m == 2) ? WvT : WoT;
  dst[n * DMODEL + k] = (_Float16)src[k * DMODEL + n];
}

// ---------------------------------------------------------------------------
// K1: LayerNorm, one wave32 per row, output f16
// ---------------------------------------------------------------------------
__global__ void layernorm_kernel(const float* __restrict__ x,
                                 const float* __restrict__ gamma,
                                 const float* __restrict__ beta,
                                 _Float16* __restrict__ xn) {
  int wave = threadIdx.x >> 5;
  int lane = threadIdx.x & 31;
  int row = blockIdx.x * (blockDim.x >> 5) + wave;
  const float* xr = x + (size_t)row * DMODEL;
  int d0 = lane * 8;
  v4f a = *(const v4f*)(xr + d0);
  v4f b = *(const v4f*)(xr + d0 + 4);
  float s  = a[0] + a[1] + a[2] + a[3] + b[0] + b[1] + b[2] + b[3];
  float sq = a[0]*a[0] + a[1]*a[1] + a[2]*a[2] + a[3]*a[3]
           + b[0]*b[0] + b[1]*b[1] + b[2]*b[2] + b[3]*b[3];
#pragma unroll
  for (int off = 16; off >= 1; off >>= 1) {
    s  += __shfl_xor(s,  off, 32);
    sq += __shfl_xor(sq, off, 32);
  }
  float mu  = s * (1.0f / DMODEL);
  float var = sq * (1.0f / DMODEL) - mu * mu;
  float rs  = rsqrtf(var + 1e-5f);
  v8h o;
#pragma unroll
  for (int i = 0; i < 8; ++i) {
    float v = (i < 4) ? a[i] : b[i - 4];
    o[i] = (_Float16)((v - mu) * rs * gamma[d0 + i] + beta[d0 + i]);
  }
  *(v8h*)(xn + (size_t)row * DMODEL + d0) = o;
}

// ---------------------------------------------------------------------------
// K2: Q/K/V projection via WMMA.  blockIdx.x = 16-row tile, blockIdx.y = proj.
// 4 waves per block, each wave owns 4 column tiles (64 cols).
// B fragments double-buffered so loads for K-step kt+1 overlap WMMAs for kt.
// Q,K stored row-major f16; V stored transposed (Vt[d][t]) f16.
// ---------------------------------------------------------------------------
__global__ void qkv_gemm_kernel(const _Float16* __restrict__ xn,
                                const _Float16* __restrict__ WqT,
                                const _Float16* __restrict__ WkT,
                                const _Float16* __restrict__ WvT,
                                const float* __restrict__ bq,
                                const float* __restrict__ bk,
                                const float* __restrict__ bv,
                                _Float16* __restrict__ Q,
                                _Float16* __restrict__ K,
                                _Float16* __restrict__ Vt) {
  int proj = blockIdx.y;
  const _Float16* WT = (proj == 0) ? WqT : (proj == 1) ? WkT : WvT;
  const float* bias  = (proj == 0) ? bq  : (proj == 1) ? bk  : bv;
  int wave = threadIdx.x >> 5;
  int lane = threadIdx.x & 31;
  int r0 = blockIdx.x * 16;
  int mrow = lane & 15;
  int mtop = (lane >> 4) * 8;

  const _Float16* arow = xn + (size_t)(r0 + mrow) * DMODEL;
  v16h afr[8];
#pragma unroll
  for (int kt = 0; kt < 8; ++kt) afr[kt] = load_frag_row(arow, kt * 32);

  const _Float16* wrow[4];
#pragma unroll
  for (int j = 0; j < 4; ++j)
    wrow[j] = WT + (size_t)((wave * 4 + j) * 16 + mrow) * DMODEL;

  v16h bcur[4];
#pragma unroll
  for (int j = 0; j < 4; ++j) bcur[j] = load_frag_row(wrow[j], 0);

  v8f acc[4] = {};
#pragma unroll
  for (int kt = 0; kt < 8; ++kt) {
    v16h bnxt[4];
    if (kt < 7) {
#pragma unroll
      for (int j = 0; j < 4; ++j) bnxt[j] = load_frag_row(wrow[j], (kt + 1) * 32);
    }
#pragma unroll
    for (int j = 0; j < 4; ++j) acc[j] = wmma16x16x32(afr[kt], bcur[j], acc[j]);
    if (kt < 7) {
#pragma unroll
      for (int j = 0; j < 4; ++j) bcur[j] = bnxt[j];
    }
  }

#pragma unroll
  for (int j = 0; j < 4; ++j) {
    int col = (wave * 4 + j) * 16 + mrow;
    float bcol = bias[col];
    if (proj < 2) {
      _Float16* dst = (proj == 0) ? Q : K;
#pragma unroll
      for (int i = 0; i < 8; ++i)
        dst[(size_t)(r0 + mtop + i) * DMODEL + col] = (_Float16)(acc[j][i] + bcol);
    } else {
      v8h o;
#pragma unroll
      for (int i = 0; i < 8; ++i) o[i] = (_Float16)(acc[j][i] + bcol);
      *(v8h*)(Vt + (size_t)col * NROWS + r0 + mtop) = o;   // contiguous t store
    }
  }
}

// ---------------------------------------------------------------------------
// K3: sliding-window attention, one wave32 per 16-query tile.
// scores (WMMA, pipelined B) -> LDS -> masked softmax -> f16 P in LDS ->
// P@V (WMMA, pipelined Vt fragments). 1/rowsum folded into the output store.
// ---------------------------------------------------------------------------
__global__ void swa_attn_kernel(const _Float16* __restrict__ Q,
                                const _Float16* __restrict__ K,
                                const _Float16* __restrict__ Vt,
                                const unsigned char* __restrict__ mask,
                                _Float16* __restrict__ O) {
  __shared__ float Sb[16 * SPAN];
  __shared__ __align__(16) _Float16 Pb[16 * SPAN];
  __shared__ float invs[16];

  int lane = threadIdx.x & 31;
  int r0 = blockIdx.x * 16;
  int b  = r0 / TSEQ;
  int t0 = r0 % TSEQ;
  int bT = b * TSEQ;
  int spanBase = r0 - HALF_W;          // global row of span index j=0
  int mrow = lane & 15;
  int mtop = (lane >> 4) * 8;
  int kh   = (lane >> 4) * 8;

  // ---- scores S = Q Kt over 160-key span ----
  const _Float16* qrow = Q + (size_t)(r0 + mrow) * DMODEL;
  v16h qfr[8];
#pragma unroll
  for (int kt = 0; kt < 8; ++kt) qfr[kt] = load_frag_row(qrow, kt * 32);

  for (int nt = 0; nt < 10; ++nt) {
    int krow = clampi(spanBase + nt * 16 + mrow, bT, bT + TSEQ - 1);
    const _Float16* krp = K + (size_t)krow * DMODEL;
    v16h bcur = load_frag_row(krp, 0);
    v8f acc = {};
#pragma unroll
    for (int kt = 0; kt < 8; ++kt) {
      v16h bnxt;
      if (kt < 7) bnxt = load_frag_row(krp, (kt + 1) * 32);
      acc = wmma16x16x32(qfr[kt], bcur, acc);
      if (kt < 7) bcur = bnxt;
    }
#pragma unroll
    for (int i = 0; i < 8; ++i)
      Sb[(mtop + i) * SPAN + nt * 16 + mrow] = acc[i];
  }
  __syncthreads();

  // ---- masked softmax: 2 lanes per row, 80 span entries each ----
  {
    int row = lane >> 1;
    int js = (lane & 1) * 80;
    const float scale = 0.0625f;  // 1/sqrt(256)
    float mx = -3.0e38f;
    for (int j = js; j < js + 80; ++j) {
      int tk = t0 - HALF_W + j;                       // key position (local)
      bool valid = (j >= row) && (j <= row + WIN - 1) // per-query window
                && (tk >= 0) && (tk < TSEQ)
                && (mask[bT + tk] != 0);
      float s = valid ? Sb[row * SPAN + j] * scale : -3.0e38f;
      Sb[row * SPAN + j] = s;
      mx = fmaxf(mx, s);
    }
    mx = fmaxf(mx, __shfl_xor(mx, 1, 32));
    float sum = 0.0f;
    for (int j = js; j < js + 80; ++j) {
      float s = Sb[row * SPAN + j];
      float e = (s > -1.0e38f) ? __expf(s - mx) : 0.0f;
      Pb[row * SPAN + j] = (_Float16)e;               // unnormalized
      sum += e;
    }
    sum += __shfl_xor(sum, 1, 32);
    if ((lane & 1) == 0) invs[row] = (sum > 0.0f) ? 1.0f / sum : 0.0f;
  }
  __syncthreads();

  // ---- O = P @ V (A fragments from LDS, B fragments from Vt rows) ----
  const _Float16* prow = &Pb[mrow * SPAN];
  v16h pfr[5];
#pragma unroll
  for (int kt = 0; kt < 5; ++kt) pfr[kt] = load_frag_row(prow, kt * 32);

  int tb0[5], tb1[5];
#pragma unroll
  for (int kt = 0; kt < 5; ++kt) {
    tb0[kt] = clampi(spanBase + kt * 32 + kh,      bT, bT + TSEQ - 8);
    tb1[kt] = clampi(spanBase + kt * 32 + kh + 16, bT, bT + TSEQ - 8);
  }

  for (int nt = 0; nt < 16; ++nt) {
    int dcol = nt * 16 + mrow;
    const _Float16* vtr = Vt + (size_t)dcol * NROWS;
    v8h lo = *(const v8h*)(vtr + tb0[0]);
    v8h hi = *(const v8h*)(vtr + tb1[0]);
    v8f acc = {};
#pragma unroll
    for (int kt = 0; kt < 5; ++kt) {
      v8h nlo, nhi;
      if (kt < 4) {
        nlo = *(const v8h*)(vtr + tb0[kt + 1]);
        nhi = *(const v8h*)(vtr + tb1[kt + 1]);
      }
      v16h bfr;
#pragma unroll
      for (int i = 0; i < 8; ++i) { bfr[i] = lo[i]; bfr[i + 8] = hi[i]; }
      acc = wmma16x16x32(pfr[kt], bfr, acc);
      if (kt < 4) { lo = nlo; hi = nhi; }
    }
#pragma unroll
    for (int i = 0; i < 8; ++i) {
      int m = mtop + i;
      O[(size_t)(r0 + m) * DMODEL + dcol] = (_Float16)(acc[i] * invs[m]);
    }
  }
}

// ---------------------------------------------------------------------------
// K4: output projection + bias + residual, f32 out (pipelined like K2)
// ---------------------------------------------------------------------------
__global__ void oproj_kernel(const _Float16* __restrict__ O,
                             const _Float16* __restrict__ WoT,
                             const float* __restrict__ bo,
                             const float* __restrict__ x,
                             float* __restrict__ out) {
  int wave = threadIdx.x >> 5;
  int lane = threadIdx.x & 31;
  int r0 = blockIdx.x * 16;
  int mrow = lane & 15;
  int mtop = (lane >> 4) * 8;

  const _Float16* arow = O + (size_t)(r0 + mrow) * DMODEL;
  v16h afr[8];
#pragma unroll
  for (int kt = 0; kt < 8; ++kt) afr[kt] = load_frag_row(arow, kt * 32);

  const _Float16* wrow[4];
#pragma unroll
  for (int j = 0; j < 4; ++j)
    wrow[j] = WoT + (size_t)((wave * 4 + j) * 16 + mrow) * DMODEL;

  v16h bcur[4];
#pragma unroll
  for (int j = 0; j < 4; ++j) bcur[j] = load_frag_row(wrow[j], 0);

  v8f acc[4] = {};
#pragma unroll
  for (int kt = 0; kt < 8; ++kt) {
    v16h bnxt[4];
    if (kt < 7) {
#pragma unroll
      for (int j = 0; j < 4; ++j) bnxt[j] = load_frag_row(wrow[j], (kt + 1) * 32);
    }
#pragma unroll
    for (int j = 0; j < 4; ++j) acc[j] = wmma16x16x32(afr[kt], bcur[j], acc[j]);
    if (kt < 7) {
#pragma unroll
      for (int j = 0; j < 4; ++j) bcur[j] = bnxt[j];
    }
  }

#pragma unroll
  for (int j = 0; j < 4; ++j) {
    int col = (wave * 4 + j) * 16 + mrow;
    float bias = bo[col];
#pragma unroll
    for (int i = 0; i < 8; ++i) {
      size_t idx = (size_t)(r0 + mtop + i) * DMODEL + col;
      out[idx] = x[idx] + acc[j][i] + bias;
    }
  }
}

// ---------------------------------------------------------------------------
extern "C" void kernel_launch(void* const* d_in, const int* in_sizes, int n_in,
                              void* d_out, int out_size, void* d_ws, size_t ws_size,
                              hipStream_t stream) {
  (void)in_sizes; (void)n_in; (void)out_size; (void)ws_size;
  const float* x   = (const float*)d_in[0];
  const unsigned char* mask = (const unsigned char*)d_in[1];  // jnp bool = 1 byte
  const float* gamma = (const float*)d_in[2];
  const float* beta  = (const float*)d_in[3];
  const float* Wq = (const float*)d_in[4];
  const float* bq = (const float*)d_in[5];
  const float* Wk = (const float*)d_in[6];
  const float* bk = (const float*)d_in[7];
  const float* Wv = (const float*)d_in[8];
  const float* bv = (const float*)d_in[9];
  const float* Wo = (const float*)d_in[10];
  const float* bo = (const float*)d_in[11];
  float* out = (float*)d_out;

  char* ws = (char*)d_ws;
  const size_t wBytes   = (size_t)DMODEL * DMODEL * sizeof(_Float16);  // 128 KB
  const size_t actBytes = (size_t)NROWS * DMODEL * sizeof(_Float16);   // 2 MB
  _Float16* WqT = (_Float16*)(ws);                 ws += wBytes;
  _Float16* WkT = (_Float16*)(ws);                 ws += wBytes;
  _Float16* WvT = (_Float16*)(ws);                 ws += wBytes;
  _Float16* WoT = (_Float16*)(ws);                 ws += wBytes;
  _Float16* xn  = (_Float16*)(ws);                 ws += actBytes;
  _Float16* Qb  = (_Float16*)(ws);                 ws += actBytes;
  _Float16* Kb  = (_Float16*)(ws);                 ws += actBytes;
  _Float16* Vt  = (_Float16*)(ws);                 ws += actBytes;
  _Float16* Ob  = (_Float16*)(ws);                 ws += actBytes;

  prep_weights_kernel<<<(4 * DMODEL * DMODEL) / 256, 256, 0, stream>>>(
      Wq, Wk, Wv, Wo, WqT, WkT, WvT, WoT);
  layernorm_kernel<<<NROWS / 4, 128, 0, stream>>>(x, gamma, beta, xn);
  qkv_gemm_kernel<<<dim3(NROWS / 16, 3), 128, 0, stream>>>(
      xn, WqT, WkT, WvT, bq, bk, bv, Qb, Kb, Vt);
  swa_attn_kernel<<<NROWS / 16, 32, 0, stream>>>(Qb, Kb, Vt, mask, Ob);
  oproj_kernel<<<NROWS / 16, 128, 0, stream>>>(Ob, WoT, bo, x, out);
}
